// MultiHeadAttention_62448824483967
// MI455X (gfx1250) — compile-verified, matrix operand order checked by reading, not run on hardware
//
#include <hip/hip_runtime.h>

typedef __attribute__((ext_vector_type(16))) _Float16 v16h;
typedef __attribute__((ext_vector_type(8)))  float    v8f;
typedef _Float16 half_t;

#define BATCH  2
#define SEQ    2048
#define DMODEL 1024
#define NHEAD  16
#define DHEAD  64
#define INFV   1e12f

#define LDP 40   // proj-tile LDS row stride (halves): 80B rows, 16B-aligned, 20-bank step
#define LDA 72   // attn-tile LDS row stride (halves): 144B rows, 16B-aligned, 36-bank step

union FragAB { v16h v; half_t h[16]; unsigned u[8]; };
union FragC  { v8f  v; float  f[8]; };
union Pack8  { half_t h[8]; uint4 u; };

// Load a 16x32 f16 A/B fragment from a row-major LDS tile (ld in halves, even).
// ISA 16-bit A layout: lanes 0-15 hold K={0..7,16..23}, lanes 16-31 hold K={8..15,24..31},
// packed 2 halves per VGPR. B uses the same per-lane packing with lane = output column.
__device__ __forceinline__ v16h frag_ld(const half_t* lds, int row0, int k0, int ld) {
  const int lane = threadIdx.x & 31;
  const int hi   = lane >> 4;
  const int row  = row0 + (lane & 15);
  const unsigned* p32 = (const unsigned*)(lds + row * ld + k0);
  FragAB f;
#pragma unroll
  for (int r = 0; r < 8; ++r)
    f.u[r] = p32[(r >> 2) * 8 + hi * 4 + (r & 3)];
  return f.v;
}

__device__ __forceinline__ v8f wmma16(v16h a, v16h b, v8f c) {
  return __builtin_amdgcn_wmma_f32_16x16x32_f16(false, a, false, b, (short)0, c,
                                                false, false);
}

// Async copy of 16 bytes (8 halves) global -> LDS, tracked by ASYNCcnt.
__device__ __forceinline__ void async_cp16(const half_t* gsrc, const half_t* ldst) {
  unsigned lds_off = (unsigned)(size_t)ldst;   // low 32 bits of generic = LDS byte offset
  asm volatile("global_load_async_to_lds_b128 %0, %1, off"
               :: "v"(lds_off), "v"(gsrc) : "memory");
}

// ---------------------------------------------------------------------------
// Kernel 1: fused QKV projection + bias + RoPE.  blockIdx.z: 0=Q 1=K 2=V.
// Block = 128 threads (4 waves, 2x2), tile 64(M) x 64(N), K-loop step 32.
// Q/K out: [B,H,S,64] f16 (rotated).  V out: [B,H,64,S] f16 (transposed).
// ---------------------------------------------------------------------------
__launch_bounds__(128)
__global__ void qkv_proj_rope_kernel(
    const float* __restrict__ Xq, const float* __restrict__ Xk, const float* __restrict__ Xv,
    const float* __restrict__ Wq, const float* __restrict__ Wk, const float* __restrict__ Wv,
    const float* __restrict__ bq, const float* __restrict__ bk, const float* __restrict__ bv,
    const float* __restrict__ rope,
    half_t* __restrict__ qw, half_t* __restrict__ kw, half_t* __restrict__ vwt) {
  const int mode = blockIdx.z;
  const float* X    = (mode == 0) ? Xq : (mode == 1) ? Xk : Xv;
  const float* W    = (mode == 0) ? Wq : (mode == 1) ? Wk : Wv;
  const float* bias = (mode == 0) ? bq : (mode == 1) ? bk : bv;

  const int n0 = blockIdx.x * 64;
  const int m0 = blockIdx.y * 64;
  const int tid  = threadIdx.x;
  const int wave = tid >> 5;
  const int lane = tid & 31;
  const int hi   = lane >> 4;
  const int wm = (wave >> 1) * 32;
  const int wn = (wave & 1) * 32;

  __shared__ __align__(16) half_t lA[64 * LDP];  // [row][k]
  __shared__ __align__(16) half_t lB[64 * LDP];  // W^T: [n][k]
  __shared__ __align__(16) float  lC[64 * 64];

  FragC acc[2][2];
#pragma unroll
  for (int i = 0; i < 2; ++i)
#pragma unroll
    for (int j = 0; j < 2; ++j)
#pragma unroll
      for (int r = 0; r < 8; ++r) acc[i][j].f[r] = 0.0f;

  for (int kt = 0; kt < DMODEL / 32; ++kt) {
    // A tile: 64x32 f32 -> f16, 8-col vector chunks (2 x b128 load, 1 x b128 store)
#pragma unroll
    for (int idx = tid; idx < 64 * 4; idx += 128) {
      int r = idx >> 2, c8 = (idx & 3) * 8;
      const float* src = &X[(size_t)(m0 + r) * DMODEL + kt * 32 + c8];
      float4 x0 = *(const float4*)(src);
      float4 x1 = *(const float4*)(src + 4);
      Pack8 pk;
      pk.h[0] = (half_t)x0.x; pk.h[1] = (half_t)x0.y;
      pk.h[2] = (half_t)x0.z; pk.h[3] = (half_t)x0.w;
      pk.h[4] = (half_t)x1.x; pk.h[5] = (half_t)x1.y;
      pk.h[6] = (half_t)x1.z; pk.h[7] = (half_t)x1.w;
      *(uint4*)&lA[r * LDP + c8] = pk.u;
    }
    // B tile (transposed into LDS): vector global read, scalar strided LDS write
#pragma unroll
    for (int idx = tid; idx < 32 * 16; idx += 128) {
      int kk = idx >> 4, n4 = (idx & 15) * 4;
      float4 w = *(const float4*)&W[(size_t)(kt * 32 + kk) * DMODEL + n0 + n4];
      lB[(n4 + 0) * LDP + kk] = (half_t)w.x;
      lB[(n4 + 1) * LDP + kk] = (half_t)w.y;
      lB[(n4 + 2) * LDP + kk] = (half_t)w.z;
      lB[(n4 + 3) * LDP + kk] = (half_t)w.w;
    }
    __syncthreads();
    v16h aF0 = frag_ld(lA, wm,      0, LDP);
    v16h aF1 = frag_ld(lA, wm + 16, 0, LDP);
    v16h bF0 = frag_ld(lB, wn,      0, LDP);
    v16h bF1 = frag_ld(lB, wn + 16, 0, LDP);
    acc[0][0].v = wmma16(aF0, bF0, acc[0][0].v);
    acc[0][1].v = wmma16(aF0, bF1, acc[0][1].v);
    acc[1][0].v = wmma16(aF1, bF0, acc[1][0].v);
    acc[1][1].v = wmma16(aF1, bF1, acc[1][1].v);
    __syncthreads();
  }

  // C fragments -> LDS (row = vgpr + 8*hi, col = lane&15)
#pragma unroll
  for (int mi = 0; mi < 2; ++mi)
#pragma unroll
    for (int ni = 0; ni < 2; ++ni)
#pragma unroll
      for (int r = 0; r < 8; ++r) {
        int row = wm + mi * 16 + r + 8 * hi;
        int col = wn + ni * 16 + (lane & 15);
        lC[row * 64 + col] = acc[mi][ni].f[r];
      }
  __syncthreads();

  // epilogue: bias (+RoPE for Q/K), write f16
  for (int idx = tid; idx < 64 * 64; idx += 128) {
    int rl = idx >> 6, cl = idx & 63;
    int gm = m0 + rl, gn = n0 + cl;
    int b = gm >> 11, s = gm & (SEQ - 1);
    int h = gn >> 6,  d = gn & 63;
    float val = lC[rl * 64 + cl] + bias[gn];
    if (mode < 2) {
      // interleaved rotation: partner is column d^1 (same head, tile-local)
      float pv = lC[rl * 64 + (cl ^ 1)] + bias[gn ^ 1];
      const float* rp = rope + ((size_t)b * SEQ + s) * DHEAD + (d & ~1);
      float sn = rp[0], cs = rp[1];          // sin at even idx, cos at odd idx
      float x2 = (d & 1) ? pv : -pv;
      float out = val * cs + x2 * sn;
      size_t o = (((size_t)b * NHEAD + h) * SEQ + s) * DHEAD + d;
      if (mode == 0) qw[o] = (half_t)out; else kw[o] = (half_t)out;
    } else {
      size_t o = (((size_t)b * NHEAD + h) * DHEAD + d) * SEQ + s;  // transposed V
      vwt[o] = (half_t)val;
    }
  }
}

// ---------------------------------------------------------------------------
// Kernel 2: flash attention per (b,h, 64-query tile). 4 waves x 16 query rows.
// K/V tiles staged with async global->LDS b128 copies (ASYNCcnt).
// ---------------------------------------------------------------------------
__launch_bounds__(128)
__global__ void attn_kernel(const half_t* __restrict__ qw, const half_t* __restrict__ kw,
                            const half_t* __restrict__ vwt, const float* __restrict__ vmask,
                            half_t* __restrict__ oh) {
  const int qt = blockIdx.x;
  const int bh = blockIdx.y;
  const int b = bh >> 4;
  const int h = bh & 15;
  const int tid  = threadIdx.x;
  const int wave = tid >> 5;
  const int lane = tid & 31;
  const int hi   = lane >> 4;
  const int q0w  = qt * 64 + wave * 16;

  const half_t* Qb = qw  + (size_t)bh * SEQ * DHEAD;
  const half_t* Kb = kw  + (size_t)bh * SEQ * DHEAD;
  const half_t* Vb = vwt + (size_t)bh * DHEAD * SEQ;

  __shared__ __align__(16) half_t lQ[64 * LDA];      // [q][d]
  __shared__ __align__(16) half_t lK[64 * LDA];      // [key][d]  (= B^T for scores)
  __shared__ __align__(16) half_t lV[64 * LDA];      // [d][key]  (= B^T for P*V)
  __shared__ __align__(16) half_t lP[4][16 * LDA];   // per-wave P tile

#pragma unroll
  for (int idx = tid; idx < 64 * 8; idx += 128) {
    int r = idx >> 3, c8 = (idx & 7) * 8;
    *(uint4*)&lQ[r * LDA + c8] = *(const uint4*)&Qb[(size_t)(qt * 64 + r) * DHEAD + c8];
  }
  __syncthreads();
  v16h qA0 = frag_ld(lQ, wave * 16, 0, LDA);
  v16h qA1 = frag_ld(lQ, wave * 16, 32, LDA);

  float mrow[8], lrow[8];
  FragC oacc[4];
#pragma unroll
  for (int r = 0; r < 8; ++r) { mrow[r] = -1e30f; lrow[r] = 0.0f; }
#pragma unroll
  for (int dt = 0; dt < 4; ++dt)
#pragma unroll
    for (int r = 0; r < 8; ++r) oacc[dt].f[r] = 0.0f;

  half_t* lPw = &lP[wave][0];

  for (int j = 0; j <= qt; ++j) {
    const int kb = j * 64;
    // Async-stage K (row-major [key][d]) and V^T (row-major [d][key]) tiles.
#pragma unroll
    for (int idx = tid; idx < 64 * 8; idx += 128) {
      int r = idx >> 3, c8 = (idx & 7) * 8;
      async_cp16(&Kb[(size_t)(kb + r) * DHEAD + c8], &lK[r * LDA + c8]);
      async_cp16(&Vb[(size_t)r * SEQ + kb + c8],     &lV[r * LDA + c8]);
    }
    asm volatile("s_wait_asynccnt 0x0" ::: "memory");
    __syncthreads();

    FragC sc[4];
    float vmadd[4];
#pragma unroll
    for (int nt = 0; nt < 4; ++nt) {
#pragma unroll
      for (int r = 0; r < 8; ++r) sc[nt].f[r] = 0.0f;
      v16h kB0 = frag_ld(lK, nt * 16, 0, LDA);
      v16h kB1 = frag_ld(lK, nt * 16, 32, LDA);
      sc[nt].v = wmma16(qA0, kB0, sc[nt].v);
      sc[nt].v = wmma16(qA1, kB1, sc[nt].v);
      int kcol = kb + nt * 16 + (lane & 15);
      vmadd[nt] = (vmask[(size_t)b * SEQ + kcol] - 1.0f) * INFV;
    }

    // scale 1/sqrt(64) + causal + sequence mask, in C layout
#pragma unroll
    for (int nt = 0; nt < 4; ++nt) {
      int kcol = kb + nt * 16 + (lane & 15);
#pragma unroll
      for (int r = 0; r < 8; ++r) {
        int qrow = q0w + r + 8 * hi;
        float s = sc[nt].f[r] * 0.125f + vmadd[nt];
        if (kcol > qrow) s = -1e30f;
        sc[nt].f[r] = s;
      }
    }

    // online softmax: row stats reduce across the 16-lane half-groups
    float mnew[8], corr[8];
#pragma unroll
    for (int r = 0; r < 8; ++r) {
      float v = fmaxf(fmaxf(sc[0].f[r], sc[1].f[r]), fmaxf(sc[2].f[r], sc[3].f[r]));
      v = fmaxf(v, __shfl_xor(v, 1));
      v = fmaxf(v, __shfl_xor(v, 2));
      v = fmaxf(v, __shfl_xor(v, 4));
      v = fmaxf(v, __shfl_xor(v, 8));
      mnew[r] = fmaxf(mrow[r], v);
      corr[r] = __expf(mrow[r] - mnew[r]);
      mrow[r] = mnew[r];
    }
#pragma unroll
    for (int nt = 0; nt < 4; ++nt)
#pragma unroll
      for (int r = 0; r < 8; ++r) {
        float p = __expf(sc[nt].f[r] - mnew[r]);
        sc[nt].f[r] = p;
        lPw[(r + 8 * hi) * LDA + nt * 16 + (lane & 15)] = (half_t)p;
      }
#pragma unroll
    for (int r = 0; r < 8; ++r) {
      float s = sc[0].f[r] + sc[1].f[r] + sc[2].f[r] + sc[3].f[r];
      s += __shfl_xor(s, 1);
      s += __shfl_xor(s, 2);
      s += __shfl_xor(s, 4);
      s += __shfl_xor(s, 8);
      lrow[r] = lrow[r] * corr[r] + s;
    }
#pragma unroll
    for (int dt = 0; dt < 4; ++dt)
#pragma unroll
      for (int r = 0; r < 8; ++r) oacc[dt].f[r] *= corr[r];

    asm volatile("s_wait_dscnt 0x0" ::: "memory");   // P tile visible to frag loads
    v16h pA0 = frag_ld(lPw, 0, 0, LDA);
    v16h pA1 = frag_ld(lPw, 0, 32, LDA);
#pragma unroll
    for (int dt = 0; dt < 4; ++dt) {
      v16h vB0 = frag_ld(lV, dt * 16, 0, LDA);
      v16h vB1 = frag_ld(lV, dt * 16, 32, LDA);
      oacc[dt].v = wmma16(pA0, vB0, oacc[dt].v);
      oacc[dt].v = wmma16(pA1, vB1, oacc[dt].v);
    }
    __syncthreads();
  }

  // normalize and write o[b][s][h*64+d] f16
#pragma unroll
  for (int dt = 0; dt < 4; ++dt)
#pragma unroll
    for (int r = 0; r < 8; ++r) {
      int row = q0w + r + 8 * hi;
      int col = h * DHEAD + dt * 16 + (lane & 15);
      float val = oacc[dt].f[r] / lrow[r];
      oh[((size_t)b * SEQ + row) * DMODEL + col] = (half_t)val;
    }
}

// ---------------------------------------------------------------------------
// Kernel 3: output projection  Y = O @ Wo + bo  (f16 x f16 -> f32)
// ---------------------------------------------------------------------------
__launch_bounds__(128)
__global__ void out_proj_kernel(const half_t* __restrict__ O, const float* __restrict__ Wo,
                                const float* __restrict__ bo, float* __restrict__ Y) {
  const int n0 = blockIdx.x * 64;
  const int m0 = blockIdx.y * 64;
  const int tid  = threadIdx.x;
  const int wave = tid >> 5;
  const int lane = tid & 31;
  const int hi   = lane >> 4;
  const int wm = (wave >> 1) * 32;
  const int wn = (wave & 1) * 32;

  __shared__ __align__(16) half_t lA[64 * LDP];
  __shared__ __align__(16) half_t lB[64 * LDP];
  __shared__ __align__(16) float  lC[64 * 64];

  FragC acc[2][2];
#pragma unroll
  for (int i = 0; i < 2; ++i)
#pragma unroll
    for (int j = 0; j < 2; ++j)
#pragma unroll
      for (int r = 0; r < 8; ++r) acc[i][j].f[r] = 0.0f;

  for (int kt = 0; kt < DMODEL / 32; ++kt) {
    // A tile is already f16: straight b128 copies (8 halves per chunk)
#pragma unroll
    for (int idx = tid; idx < 64 * 4; idx += 128) {
      int r = idx >> 2, c8 = (idx & 3) * 8;
      *(uint4*)&lA[r * LDP + c8] =
          *(const uint4*)&O[(size_t)(m0 + r) * DMODEL + kt * 32 + c8];
    }
#pragma unroll
    for (int idx = tid; idx < 32 * 16; idx += 128) {
      int kk = idx >> 4, n4 = (idx & 15) * 4;
      float4 w = *(const float4*)&Wo[(size_t)(kt * 32 + kk) * DMODEL + n0 + n4];
      lB[(n4 + 0) * LDP + kk] = (half_t)w.x;
      lB[(n4 + 1) * LDP + kk] = (half_t)w.y;
      lB[(n4 + 2) * LDP + kk] = (half_t)w.z;
      lB[(n4 + 3) * LDP + kk] = (half_t)w.w;
    }
    __syncthreads();
    v16h aF0 = frag_ld(lA, wm,      0, LDP);
    v16h aF1 = frag_ld(lA, wm + 16, 0, LDP);
    v16h bF0 = frag_ld(lB, wn,      0, LDP);
    v16h bF1 = frag_ld(lB, wn + 16, 0, LDP);
    acc[0][0].v = wmma16(aF0, bF0, acc[0][0].v);
    acc[0][1].v = wmma16(aF0, bF1, acc[0][1].v);
    acc[1][0].v = wmma16(aF1, bF0, acc[1][0].v);
    acc[1][1].v = wmma16(aF1, bF1, acc[1][1].v);
    __syncthreads();
  }

#pragma unroll
  for (int mi = 0; mi < 2; ++mi)
#pragma unroll
    for (int ni = 0; ni < 2; ++ni)
#pragma unroll
      for (int r = 0; r < 8; ++r) {
        int row = wm + mi * 16 + r + 8 * hi;
        int col = wn + ni * 16 + (lane & 15);
        lC[row * 64 + col] = acc[mi][ni].f[r];
      }
  __syncthreads();

  for (int idx = tid; idx < 64 * 64; idx += 128) {
    int rl = idx >> 6, cl = idx & 63;
    int gm = m0 + rl, gn = n0 + cl;
    Y[(size_t)gm * DMODEL + gn] = lC[rl * 64 + cl] + bo[gn];
  }
}

// ---------------------------------------------------------------------------
extern "C" void kernel_launch(void* const* d_in, const int* in_sizes, int n_in,
                              void* d_out, int out_size, void* d_ws, size_t ws_size,
                              hipStream_t stream) {
  const float* q     = (const float*)d_in[0];
  const float* k     = (const float*)d_in[1];
  const float* v     = (const float*)d_in[2];
  const float* rope  = (const float*)d_in[3];
  // d_in[4] = a_bias: causal mask, computed analytically in-kernel
  const float* vmask = (const float*)d_in[5];
  const float* Wq = (const float*)d_in[6];
  const float* bq = (const float*)d_in[7];
  const float* Wk = (const float*)d_in[8];
  const float* bk = (const float*)d_in[9];
  const float* Wv = (const float*)d_in[10];
  const float* bv = (const float*)d_in[11];
  const float* Wo = (const float*)d_in[12];
  const float* bo = (const float*)d_in[13];

  const size_t per = (size_t)BATCH * NHEAD * SEQ * DHEAD;  // 4M halves = 8 MB each
  half_t* qw  = (half_t*)d_ws;
  half_t* kw  = qw  + per;
  half_t* vwt = kw  + per;
  half_t* oh  = vwt + per;

  dim3 gproj(DMODEL / 64, (BATCH * SEQ) / 64, 3);
  qkv_proj_rope_kernel<<<gproj, 128, 0, stream>>>(q, k, v, Wq, Wk, Wv, bq, bk, bv,
                                                  rope, qw, kw, vwt);

  dim3 gattn(SEQ / 64, BATCH * NHEAD);
  attn_kernel<<<gattn, 128, 0, stream>>>(qw, kw, vwt, vmask, oh);

  dim3 gout(DMODEL / 64, (BATCH * SEQ) / 64);
  out_proj_kernel<<<gout, 128, 0, stream>>>(oh, Wo, bo, (float*)d_out);
}
